// DualSTGCN_w_EHR_61065845014840
// MI455X (gfx1250) — matrix operand order repaired and checked by reading, not compiled, not run on hardware
//
// DualSTGCN_w_EHR on MI455X (gfx1250, wave32, WMMA).
//
// Optimization: the conv1d -> ChebConv(ring) -> pooling chain is linear, so it
// is folded on-device into one effective GEMM per branch:
//   ecc_g = X_ecc[B,400] @ Meff_ecc[400,256] + beff_ecc
//   err_g = X_err[B,300] @ Meff_err[300,256] + beff_err
// (ring L_hat = -0.5*A, neighbor mixing folded into Meff rows per node).
// This drops FLOPs from ~56 GFLOP to ~3 GFLOP and, critically, avoids
// materializing the 366MB temporal-conv expansion: the whole net becomes
// HBM-bound on the 25MB of inputs (~1.1us at 23.3 TB/s). GEMMs run on
// v_wmma_f32_16x16x32_f16 (f16 in, f32 accumulate), weights pre-swizzled into
// WMMA B-fragment order so the hot loop does pure coalesced 32B loads from L2.

#include <hip/hip_runtime.h>
#include <hip/hip_bf16.h>

typedef __attribute__((ext_vector_type(16))) _Float16 v16h;
typedef __attribute__((ext_vector_type(8)))  float    v8f;
typedef __attribute__((ext_vector_type(4)))  float    v4f;

// ---------------- prep 1: fold conv1d into Cheb weights ---------------------
// W0eff[t',g] = sum_c sum_k wt[c,0,k] * W0[(c*25 + t'-k+1), g]   (SAME pad)
// cb[g] = b_branch[g] + sum_f bt[c(f)]*(W0[f,g] - W1[f,g])       (L_hat*1 = -1)
__global__ __launch_bounds__(256) void prep1_fold_conv(
    const float* __restrict__ wt_ecc, const float* __restrict__ bt_ecc,
    const float* __restrict__ wt_err, const float* __restrict__ bt_err,
    const float* __restrict__ W0e, const float* __restrict__ W1e, const float* __restrict__ bbe,
    const float* __restrict__ W0r, const float* __restrict__ W1r, const float* __restrict__ bbr,
    float* __restrict__ aeff_ecc, float* __restrict__ aeff_err,
    float* __restrict__ cb_ecc, float* __restrict__ cb_err)
{
    int idx = blockIdx.x * 256 + threadIdx.x;
    const int NW = 25 * 64;
    if (idx < 4 * NW) {
        int which = idx / NW;            // 0:W0ecc 1:W1ecc 2:W0err 3:W1err
        int r = idx % NW;
        int t = r >> 6, g = r & 63;
        const float* wt = (which < 2) ? wt_ecc : wt_err;
        const float* W  = (which == 0) ? W0e : (which == 1) ? W1e : (which == 2) ? W0r : W1r;
        float s = 0.f;
        for (int c = 0; c < 32; ++c)
            for (int k = 0; k < 3; ++k) {
                int tt = t - k + 1;
                if (tt >= 0 && tt < 25) s += wt[c * 3 + k] * W[(c * 25 + tt) * 64 + g];
            }
        float* dst = (which < 2) ? aeff_ecc : aeff_err;
        dst[(which & 1) * 1600 + t * 64 + g] = s;
    } else if (idx < 4 * NW + 128) {
        int r = idx - 4 * NW;
        int br = r >> 6, g = r & 63;
        const float* bt = br ? bt_err : bt_ecc;
        const float* W0 = br ? W0r : W0e;
        const float* W1 = br ? W1r : W1e;
        const float* bb = br ? bbr : bbe;
        float s = bb[g];
        for (int f = 0; f < 800; ++f)
            s += bt[f / 25] * (W0[f * 64 + g] - W1[f * 64 + g]);
        (br ? cb_err : cb_ecc)[g] = s;
    }
}

// ---------------- prep 2: fold Cheb + ring mixing into pooling GEMM ---------
// Meff[u*25+t, h] = sum_g W0eff[t,g]*Wp[u*64+g, h]
//                 - 0.5*W1eff[t,g]*(Wp[(u+1)%V...] + Wp[(u-1)%V...])
// beff[h] = bp[h] + sum_{u,g} cb[g]*Wp[u*64+g, h]
__global__ __launch_bounds__(256) void prep2_build_meff(
    const float* __restrict__ aeff_ecc, const float* __restrict__ aeff_err,
    const float* __restrict__ cb_ecc,   const float* __restrict__ cb_err,
    const float* __restrict__ Wpe, const float* __restrict__ bpe,
    const float* __restrict__ Wpr, const float* __restrict__ bpr,
    float* __restrict__ meff_ecc, float* __restrict__ meff_err,
    float* __restrict__ beff_ecc, float* __restrict__ beff_err)
{
    int idx = blockIdx.x * 256 + threadIdx.x;
    const int NE = 400 * 256, NR = 300 * 256;
    if (idx < NE + NR) {
        int br = (idx >= NE);
        int r  = br ? idx - NE : idx;
        const int V = br ? 12 : 16;
        const float* A  = br ? aeff_err : aeff_ecc;
        const float* Wp = br ? Wpr : Wpe;
        int row = r >> 8, h = r & 255;
        int u = row / 25, t = row % 25;
        int up = (u + 1) % V, um = (u + V - 1) % V;
        float s = 0.f;
        for (int g = 0; g < 64; ++g) {
            float w0 = A[t * 64 + g];
            float w1 = A[1600 + t * 64 + g];
            s += w0 * Wp[(u  * 64 + g) * 256 + h]
               - 0.5f * w1 * (Wp[(up * 64 + g) * 256 + h] + Wp[(um * 64 + g) * 256 + h]);
        }
        (br ? meff_err : meff_ecc)[row * 256 + h] = s;
    } else if (idx < NE + NR + 512) {
        int r = idx - NE - NR;
        int br = (r >= 256);
        int h = r & 255;
        const int V = br ? 12 : 16;
        const float* cb = br ? cb_err : cb_ecc;
        const float* Wp = br ? Wpr : Wpe;
        float s = (br ? bpr : bpe)[h];
        for (int u = 0; u < V; ++u)
            for (int g = 0; g < 64; ++g)
                s += cb[g] * Wp[(u * 64 + g) * 256 + h];
        (br ? beff_err : beff_ecc)[h] = s;
    }
}

// ---------------- prep 3: swizzle weights to f16 WMMA B-fragment order ------
// B (32x16 f16, 8 VGPRs): lane l -> column N = l&15; e=0..7 at K=(l>>4)*8+e, e=8..15 at K=16+(l>>4)*8+(e-8).
// Stored as [kstep][ntile][lane][16 halves] so hot loop loads 32B/lane coalesced.
__global__ __launch_bounds__(256) void prep3_swizzle(
    const float* __restrict__ meff_ecc, const float* __restrict__ meff_err,
    const float* __restrict__ We,
    _Float16* __restrict__ bfe, _Float16* __restrict__ bfr, _Float16* __restrict__ bfw)
{
    int idx = blockIdx.x * 256 + threadIdx.x;
    const int HE = 13 * 16 * 512, HR = 10 * 16 * 512, HW = 2 * 4 * 512;
    const float* src; _Float16* dst; int NT, Kv, N, o;
    if (idx < HE)                { src = meff_ecc; dst = bfe; NT = 16; Kv = 400; N = 256; o = idx; }
    else if (idx < HE + HR)      { src = meff_err; dst = bfr; NT = 16; Kv = 300; N = 256; o = idx - HE; }
    else if (idx < HE + HR + HW) { src = We;       dst = bfw; NT = 4;  Kv = 64;  N = 64;  o = idx - HE - HR; }
    else return;
    int s   = o / (NT * 512);
    int rem = o % (NT * 512);
    int n = rem / 512, q = rem % 512;
    int lane = q >> 4, e = q & 15;
    int K    = s * 32 + ((lane >> 4) << 3) + (e & 7) + ((e >> 3) << 4);
    int ncol = n * 16 + (lane & 15);
    float v = (K < Kv) ? src[K * N + ncol] : 0.f;
    dst[o] = (_Float16)v;
}

// ---------------- main fused kernel -----------------------------------------
// Load 8 consecutive f32 from a row (with K-bounds guard), convert to f16.
__device__ inline void load8_cvt(_Float16* dst, const float* __restrict__ row, int k, int kv)
{
    if (k + 8 <= kv) {
        v4f a = *(const v4f*)(row + k);
        v4f b = *(const v4f*)(row + k + 4);
#pragma unroll
        for (int i = 0; i < 4; ++i) { dst[i] = (_Float16)a[i]; dst[4 + i] = (_Float16)b[i]; }
    } else {
#pragma unroll
        for (int i = 0; i < 8; ++i) dst[i] = (k + i < kv) ? (_Float16)row[k + i] : (_Float16)0.f;
    }
}

// A (16x32 f16): lane l -> row M = l&15; halves e=0..7 at K=(l>>4)*8+e,
// e=8..15 at K=16+(l>>4)*8+(e-8).  acc = D/C (16x16 f32, 8 VGPRs).
template <int NS, int NT, int KV>
__device__ inline void gemm_rows16(const float* __restrict__ xrow,
                                   const _Float16* __restrict__ bfrag,
                                   int lane, v8f* acc)
{
    const int base0 = (lane >> 4) << 3;
    for (int s = 0; s < NS; ++s) {
        union { v16h v; _Float16 e[16]; } af;
        int k0 = s * 32 + base0;
        load8_cvt(af.e,     xrow, k0,      KV);
        load8_cvt(af.e + 8, xrow, k0 + 16, KV);
#pragma unroll
        for (int n = 0; n < NT; ++n) {
            v16h bf = *(const v16h*)(bfrag + (((size_t)(s * NT + n) * 32 + lane) << 4));
            acc[n] = __builtin_amdgcn_wmma_f32_16x16x32_f16(
                false, af.v, false, bf, (short)0, acc[n], false, false);
        }
    }
}

__global__ __launch_bounds__(32) void fused_dualstgcn(
    const float* __restrict__ ecc, const float* __restrict__ err, const float* __restrict__ ehr,
    const _Float16* __restrict__ bfe, const _Float16* __restrict__ bfr, const _Float16* __restrict__ bfw,
    const float* __restrict__ beffe, const float* __restrict__ beffr,
    const float* __restrict__ Wa, const float* __restrict__ ba,
    const float* __restrict__ be, const float* __restrict__ Wf2, const float* __restrict__ bf2,
    float* __restrict__ out)
{
    __shared__ float sm_eg[16 * 256];
    __shared__ float sm_rg[16 * 256];
    __shared__ float sm_hp[16 * 64];

    const int l = threadIdx.x;            // 0..31 (wave32)
    const int rowBase = blockIdx.x << 4;  // 16 rows per wave, 512 blocks = 8192
    const int mrow = l & 15;
    const int rhi  = (l >> 4) << 3;       // D row offset: lanes 16-31 hold M+8

    v8f acc[16];

    // ---- ecc_g = X_ecc @ Meff_ecc  (K=400 padded to 416, 13 ksteps, 16 ntiles)
    for (int n = 0; n < 16; ++n) for (int i = 0; i < 8; ++i) acc[n][i] = 0.f;
    gemm_rows16<13, 16, 400>(ecc + (size_t)(rowBase + mrow) * 400, bfe, l, acc);
    for (int n = 0; n < 16; ++n)
        for (int r = 0; r < 8; ++r)
            sm_eg[(r + rhi) * 256 + n * 16 + mrow] = acc[n][r];

    // ---- err_g = X_err @ Meff_err  (K=300 padded to 320, 10 ksteps)
    for (int n = 0; n < 16; ++n) for (int i = 0; i < 8; ++i) acc[n][i] = 0.f;
    gemm_rows16<10, 16, 300>(err + (size_t)(rowBase + mrow) * 300, bfr, l, acc);
    for (int n = 0; n < 16; ++n)
        for (int r = 0; r < 8; ++r)
            sm_rg[(r + rhi) * 256 + n * 16 + mrow] = acc[n][r];

    // ---- ehr_p = relu(ehr @ We + be)  (K=64, 2 ksteps, 4 ntiles)
    v8f a2[4];
    for (int n = 0; n < 4; ++n) for (int i = 0; i < 8; ++i) a2[n][i] = 0.f;
    gemm_rows16<2, 4, 64>(ehr + (size_t)(rowBase + mrow) * 64, bfw, l, a2);
    for (int n = 0; n < 4; ++n)
        for (int r = 0; r < 8; ++r) {
            int col = n * 16 + mrow;
            float v = a2[n][r] + be[col];
            sm_hp[(r + rhi) * 64 + col] = v > 0.f ? v : 0.f;
        }

    __syncthreads();

    // ---- attention + fusion + head (wave32 reductions)
    float wa[8], wf[8], bea[8], ber[8];
#pragma unroll
    for (int j = 0; j < 8; ++j) {
        int h = l + j * 32;
        wa[j] = Wa[h]; wf[j] = Wf2[h]; bea[j] = beffe[h]; ber[j] = beffr[h];
    }
    float wfh0 = Wf2[256 + l], wfh1 = Wf2[288 + l];
    float bav = ba[0], bfv = bf2[0];

    for (int r = 0; r < 16; ++r) {
        float ev[8], gv[8], t = 0.f;
#pragma unroll
        for (int j = 0; j < 8; ++j) {
            int h = l + j * 32;
            ev[j] = sm_eg[r * 256 + h] + bea[j];
            gv[j] = sm_rg[r * 256 + h] + ber[j];
            t += tanhf(ev[j] + gv[j]) * wa[j];
        }
        for (int o = 16; o; o >>= 1) t += __shfl_xor(t, o, 32);
        float attn = 1.f / (1.f + expf(-(t + bav)));

        float sacc = 0.f;
#pragma unroll
        for (int j = 0; j < 8; ++j) {
            float f = attn * ev[j] + (1.f - attn) * gv[j];
            sacc += (f > 0.f ? f : 0.f) * wf[j];
        }
        sacc += sm_hp[r * 64 + l] * wfh0 + sm_hp[r * 64 + 32 + l] * wfh1;
        for (int o = 16; o; o >>= 1) sacc += __shfl_xor(sacc, o, 32);

        if (l == 0) out[rowBase + r] = 1.f / (1.f + expf(-(sacc + bfv)));
    }
}

// ---------------- host launcher ---------------------------------------------
extern "C" void kernel_launch(void* const* d_in, const int* in_sizes, int n_in,
                              void* d_out, int out_size, void* d_ws, size_t ws_size,
                              hipStream_t stream)
{
    (void)in_sizes; (void)n_in; (void)out_size; (void)ws_size;

    const float* ecc = (const float*)d_in[0];
    const float* err = (const float*)d_in[1];
    const float* ehr = (const float*)d_in[2];
    const float* wt_ecc = (const float*)d_in[3];
    const float* bt_ecc = (const float*)d_in[4];
    const float* wt_err = (const float*)d_in[5];
    const float* bt_err = (const float*)d_in[6];
    const float* W0e = (const float*)d_in[7];
    const float* W1e = (const float*)d_in[8];
    const float* bbe = (const float*)d_in[9];
    const float* W0r = (const float*)d_in[10];
    const float* W1r = (const float*)d_in[11];
    const float* bbr = (const float*)d_in[12];
    const float* Wpe = (const float*)d_in[13];
    const float* bpe = (const float*)d_in[14];
    const float* Wpr = (const float*)d_in[15];
    const float* bpr = (const float*)d_in[16];
    const float* Wa  = (const float*)d_in[17];
    const float* ba  = (const float*)d_in[18];
    const float* We  = (const float*)d_in[19];
    const float* be  = (const float*)d_in[20];
    const float* Wf2 = (const float*)d_in[21];
    const float* bf2 = (const float*)d_in[22];

    // workspace carve-up (f32 region then f16 region; offsets keep 32B align)
    float* wf        = (float*)d_ws;
    float* aeff_ecc  = wf;                    // 3200
    float* aeff_err  = aeff_ecc + 3200;       // 3200
    float* cb_ecc    = aeff_err + 3200;       // 64
    float* cb_err    = cb_ecc + 64;           // 64
    float* meff_ecc  = cb_err + 64;           // 400*256
    float* meff_err  = meff_ecc + 400 * 256;  // 300*256
    float* beff_ecc  = meff_err + 300 * 256;  // 256
    float* beff_err  = beff_ecc + 256;        // 256  -> 186240 floats total
    _Float16* bfe = (_Float16*)((char*)d_ws + (size_t)186240 * sizeof(float));
    _Float16* bfr = bfe + 13 * 16 * 512;      // 106496 halves
    _Float16* bfw = bfr + 10 * 16 * 512;      // 81920 halves (+4096 for We)

    prep1_fold_conv<<<26, 256, 0, stream>>>(
        wt_ecc, bt_ecc, wt_err, bt_err, W0e, W1e, bbe, W0r, W1r, bbr,
        aeff_ecc, aeff_err, cb_ecc, cb_err);

    prep2_build_meff<<<(400 * 256 + 300 * 256 + 512 + 255) / 256, 256, 0, stream>>>(
        aeff_ecc, aeff_err, cb_ecc, cb_err, Wpe, bpe, Wpr, bpr,
        meff_ecc, meff_err, beff_ecc, beff_err);

    prep3_swizzle<<<(13 * 16 * 512 + 10 * 16 * 512 + 2 * 4 * 512 + 255) / 256, 256, 0, stream>>>(
        meff_ecc, meff_err, We, bfe, bfr, bfw);

    fused_dualstgcn<<<512, 32, 0, stream>>>(
        ecc, err, ehr, bfe, bfr, bfw, beff_ecc, beff_err,
        Wa, ba, be, Wf2, bf2, (float*)d_out);
}
